// Transformer_16638703305171
// MI455X (gfx1250) — matrix operand order repaired and checked by reading, not compiled, not run on hardware
//
#include <hip/hip_runtime.h>
#include <math.h>

typedef _Float16 f16;
typedef __attribute__((ext_vector_type(16))) _Float16 v16h;
typedef __attribute__((ext_vector_type(8)))  float    v8f;

#define B_     4
#define BPTT_  16
#define PRED_  4
#define S_     20
#define V_     512      // 8*8*8 voxels
#define E_     256
#define NH_    8
#define HEAD_  32
#define HID_   512
#define NE_    512
#define L_     4
#define NTOK_  (B_*S_*V_)      // 40960 rows
#define NMASK_ (B_*PRED_*V_)   // 8192 masked rows

// ---------------------------------------------------------------------------
// Weight pre-pack: f32 [ntaps][Cin][Cout] -> f16 WMMA B-fragments.
// Fragment f = ((t*Cin/32 + kc)*Cout/16 + nt); lane l holds 16 halves (32B
// contiguous): VGPR i = K { (l/16)*16 + 2i, +1 } of column n = nt*16 + (l%16)
// (contiguous k order, not the ISA table order; A loads use the same order).
// ---------------------------------------------------------------------------
__global__ void pack_w_k(const float* __restrict__ w, f16* __restrict__ wp,
                         int Cin, int Cout, int ntaps) {
  long nd = (long)ntaps * Cin * Cout / 2;            // packed dwords
  long d = (long)blockIdx.x * blockDim.x + threadIdx.x;
  if (d >= nd) return;
  int f    = (int)(d >> 8);                          // 256 dwords / fragment
  int r    = (int)(d & 255);
  int lane = r >> 3;
  int i    = r & 7;
  int ntPer = Cout >> 4, kcPer = Cin >> 5;
  int nt = f % ntPer;
  int kc = (f / ntPer) % kcPer;
  int t  = f / (ntPer * kcPer);
  int k  = kc * 32 + (lane >> 4) * 16 + 2 * i;
  int n  = nt * 16 + (lane & 15);
  float a = w[((long)t * Cin + k) * Cout + n];
  float b = w[((long)t * Cin + k + 1) * Cout + n];
  union { f16 h[2]; unsigned u; } pk;
  pk.h[0] = (f16)a; pk.h[1] = (f16)b;
  ((unsigned*)wp)[d] = pk.u;
}

// Fused QKV pre-pack: three f32 [27][256][256] arrays -> one fragment stream
// with logical Cout = 768 (columns 0..255 = Q, 256..511 = K, 512..767 = V).
__global__ void pack_qkv_k(const float* __restrict__ wq, const float* __restrict__ wk,
                           const float* __restrict__ wv, f16* __restrict__ wp) {
  const int Cin = 256, Cout = 768, ntaps = 27;
  long nd = (long)ntaps * Cin * Cout / 2;
  long d = (long)blockIdx.x * blockDim.x + threadIdx.x;
  if (d >= nd) return;
  int f    = (int)(d >> 8);
  int r    = (int)(d & 255);
  int lane = r >> 3;
  int i    = r & 7;
  int ntPer = Cout >> 4, kcPer = Cin >> 5;
  int nt = f % ntPer;
  int kc = (f / ntPer) % kcPer;
  int t  = f / (ntPer * kcPer);
  int k  = kc * 32 + (lane >> 4) * 16 + 2 * i;
  int n  = nt * 16 + (lane & 15);
  const float* w = (n < 256) ? wq : (n < 512) ? wk : wv;
  int nn = n & 255;
  float a = w[((long)t * Cin + k) * 256 + nn];
  float b = w[((long)t * Cin + k + 1) * 256 + nn];
  union { f16 h[2]; unsigned u; } pk;
  pk.h[0] = (f16)a; pk.h[1] = (f16)b;
  ((unsigned*)wp)[d] = pk.u;
}

__global__ void concat_bias_k(const float* __restrict__ a, const float* __restrict__ b,
                              const float* __restrict__ c, float* __restrict__ o) {
  int i = blockIdx.x * blockDim.x + threadIdx.x;
  if (i < 256) o[i] = a[i];
  else if (i < 512) o[i] = b[i - 256];
  else if (i < 768) o[i] = c[i - 512];
}

// ---------------------------------------------------------------------------
// Implicit-GEMM conv3d (SAME 3x3x3, or 1x1x1) via WMMA f16 -> f32.
// Block: 128 threads = 4 waves; tile 128(M) x 64(N); each wave owns 32Mx64N
// (two A fragments share every B fragment -> 8 WMMAs per K-step per wave).
// A-tile staging uses GLOBAL_LOAD_ASYNC_TO_LDS_B128 (ASYNCcnt) so the copy
// never touches VGPRs; padding rows are zero-filled with plain LDS stores.
// map_mode==1 remaps dense image g -> (g/PRED)*S + BPTT + g%PRED (decoder
// runs only on the PRED masked frames). qkv_split!=0 scatters columns of a
// fused Cout=768 GEMM into three contiguous [M][256] f16 buffers.
// ---------------------------------------------------------------------------
__launch_bounds__(128)
__global__ void conv_wmma_k(const f16* __restrict__ in, const f16* __restrict__ wp,
                            const float* __restrict__ bias,
                            float* __restrict__ out32, f16* __restrict__ out16,
                            float* __restrict__ resid,
                            int Cin, int Cout, int ntaps, int M,
                            int map_mode, int do_gelu, long qkv_split) {
  const int nTileN = Cout >> 6;
  const int bm = blockIdx.x / nTileN;
  const int bn = blockIdx.x % nTileN;
  const int m0 = bm << 7;
  const int n0 = bn << 6;
  const int tid  = threadIdx.x;
  const int wave = tid >> 5;
  const int lane = tid & 31;
  const int kcN    = Cin >> 5;
  const int nFragC = Cout >> 4;

  // 128 rows x 32 halves, stride 40 halves (80B = 5*16B: 16B-aligned rows,
  // and 20*r mod 64 gives conflict-free banks across the 16 row-lanes).
  __shared__ f16 As[128 * 40];

  v8f acc[2][4] = {};

  // staging: one thread per row, 64B contiguous per K-step
  const int sr = tid;
  int mRow = m0 + sr;
  int img  = mRow >> 9;
  int vox  = mRow & 511;
  if (map_mode == 1) img = (img >> 2) * S_ + BPTT_ + (img & 3);
  const int vx = vox >> 6, vy = (vox >> 3) & 7, vz = vox & 7;

  // low 32 bits of the generic shared address == LDS byte offset (ISA 10.2)
  const unsigned ldsRow = (unsigned)(size_t)(&As[sr * 40]);

  const int mf0 = (wave << 5) + (lane & 15);   // first A-fragment row
  const int hs  = lane >> 4;

  for (int t = 0; t < ntaps; ++t) {
    int tx = 0, ty = 0, tz = 0;
    if (ntaps == 27) { tz = t % 3 - 1; ty = (t / 3) % 3 - 1; tx = t / 9 - 1; }
    const int nx = vx + tx, ny = vy + ty, nz = vz + tz;
    const bool ok = ((unsigned)nx < 8u) && ((unsigned)ny < 8u) && ((unsigned)nz < 8u);
    const f16* src = in + (((long)img * V_ + ((nx * 8 + ny) * 8 + nz)) * Cin);

    for (int kc = 0; kc < kcN; ++kc) {
      __syncthreads();
      if (ok) {
        // async copy 64B: instruction offset is added to BOTH the LDS and
        // global addresses (ISA 10.4.4), so 4 x b128 moves the whole row slice
        unsigned long long ga = (unsigned long long)(size_t)(src + kc * 32);
        asm volatile(
            "global_load_async_to_lds_b128 %0, %1, off\n\t"
            "global_load_async_to_lds_b128 %0, %1, off offset:16\n\t"
            "global_load_async_to_lds_b128 %0, %1, off offset:32\n\t"
            "global_load_async_to_lds_b128 %0, %1, off offset:48"
            :: "v"(ldsRow), "v"(ga) : "memory");
      } else {
        uint4 z = make_uint4(0u, 0u, 0u, 0u);
        uint4* dst = (uint4*)&As[sr * 40];
        dst[0] = z; dst[1] = z; dst[2] = z; dst[3] = z;
      }
      asm volatile("s_wait_asynccnt 0x0" ::: "memory");
      __syncthreads();

      // Two A fragments (contiguous k order, same as the packed B fragments)
      union { v16h v; unsigned u[8]; } Af[2];
      #pragma unroll
      for (int i = 0; i < 8; ++i) {
        int kb = hs * 16 + 2 * i;
        Af[0].u[i] = *(const unsigned*)&As[mf0 * 40 + kb];
        Af[1].u[i] = *(const unsigned*)&As[(mf0 + 16) * 40 + kb];
      }

      const long fbase = ((long)(t * kcN + kc)) * nFragC + (n0 >> 4);
      __builtin_prefetch(wp + (fbase + nFragC) * 512, 0, 1);   // next K-step B frags

      #pragma unroll
      for (int j = 0; j < 4; ++j) {
        union { v16h v; uint4 q[2]; } Bf;
        const uint4* bp = (const uint4*)(wp + (fbase + j) * 512 + lane * 16);
        Bf.q[0] = bp[0]; Bf.q[1] = bp[1];
        acc[0][j] = __builtin_amdgcn_wmma_f32_16x16x32_f16(
            false, Af[0].v, false, Bf.v, (short)0, acc[0][j], false, false);
        acc[1][j] = __builtin_amdgcn_wmma_f32_16x16x32_f16(
            false, Af[1].v, false, Bf.v, (short)0, acc[1][j], false, false);
      }
    }
  }

  // Epilogue: C/D f32 16x16 layout -> (m, n); bias + GELU + residual + stores
  const int nl = lane & 15;
  #pragma unroll
  for (int a2 = 0; a2 < 2; ++a2) {
    #pragma unroll
    for (int j = 0; j < 4; ++j) {
      const int n = n0 + j * 16 + nl;
      const float bv = bias ? bias[n] : 0.0f;
      #pragma unroll
      for (int r = 0; r < 8; ++r) {
        const int m = m0 + (wave << 5) + (a2 << 4) + r + (hs << 3);
        float v = acc[a2][j][r] + bv;
        if (do_gelu) v = 0.5f * v * (1.0f + erff(v * 0.70710678118654752f));
        if (qkv_split) {
          out16[(long)(n >> 8) * qkv_split + (long)m * 256 + (n & 255)] = (f16)v;
        } else {
          const long idx = (long)m * Cout + n;
          if (resid) { v += resid[idx]; resid[idx] = v; }
          if (out32) out32[idx] = v;
          if (out16) out16[idx] = (f16)v;
        }
      }
    }
  }
}

// ---------------------------------------------------------------------------
// Embedding gather + positional add: X[b,s,vox,e] (f32 residual stream)
// ---------------------------------------------------------------------------
__global__ void embed_k(const int* __restrict__ code, const float* __restrict__ emb,
                        const float* __restrict__ pos, float* __restrict__ X) {
  long gid = (long)blockIdx.x * blockDim.x + threadIdx.x;
  if (gid >= (long)NTOK_ * E_) return;
  int e = (int)(gid & (E_ - 1));
  long row = gid >> 8;
  int vox = (int)(row % V_);
  int s   = (int)((row / V_) % S_);
  int b   = (int)(row / ((long)V_ * S_));
  int tok = (s < BPTT_) ? code[((long)b * BPTT_ + s) * V_ + vox] : NE_;
  X[gid] = emb[(long)tok * E_ + e] + pos[(long)s * E_ + e];
}

// ---------------------------------------------------------------------------
// LayerNorm: one wave32 per row, f32 in -> f16 out
// ---------------------------------------------------------------------------
__global__ void layernorm_k(const float* __restrict__ X, const float* __restrict__ g,
                            const float* __restrict__ be, f16* __restrict__ out,
                            int rows, int C) {
  int row  = blockIdx.x * (blockDim.x >> 5) + (threadIdx.x >> 5);
  int lane = threadIdx.x & 31;
  if (row >= rows) return;
  const float* xr = X + (long)row * C;
  float s = 0.f, s2 = 0.f;
  for (int c = lane; c < C; c += 32) { float v = xr[c]; s += v; s2 += v * v; }
  #pragma unroll
  for (int o = 16; o; o >>= 1) { s += __shfl_xor(s, o, 32); s2 += __shfl_xor(s2, o, 32); }
  float m   = s / C;
  float var = s2 / C - m * m;
  float inv = rsqrtf(var + 1e-5f);
  for (int c = lane; c < C; c += 32)
    out[(long)row * C + c] = (f16)((xr[c] - m) * inv * g[c] + be[c]);
}

// ---------------------------------------------------------------------------
// Attention over time (S=20) per (b, voxel, head, s): scalar path (tiny FLOPs)
// Q/K/V are contiguous [M][E] f16 planes inside one buffer.
// ---------------------------------------------------------------------------
__global__ void attention_k(const f16* __restrict__ QKV, long plane,
                            f16* __restrict__ O) {
  long gid = (long)blockIdx.x * blockDim.x + threadIdx.x;
  if (gid >= (long)B_ * V_ * NH_ * S_) return;
  int s   = (int)(gid % S_);
  int h   = (int)((gid / S_) % NH_);
  int vox = (int)((gid / ((long)S_ * NH_)) % V_);
  int b   = (int)(gid / ((long)S_ * NH_ * V_));
  const f16* Q  = QKV;
  const f16* K  = QKV + plane;
  const f16* Vv = QKV + 2 * plane;

  float q[HEAD_];
  const f16* qp = Q + ((((long)b * S_ + s) * V_ + vox) * E_ + h * HEAD_);
  #pragma unroll
  for (int d = 0; d < HEAD_; ++d) q[d] = (float)qp[d];

  float sc[S_];
  float mx = -3.4e38f;
  for (int t = 0; t < S_; ++t) {
    const f16* kp = K + ((((long)b * S_ + t) * V_ + vox) * E_ + h * HEAD_);
    float a = 0.f;
    #pragma unroll
    for (int d = 0; d < HEAD_; ++d) a += q[d] * (float)kp[d];
    a *= 0.17677669529663687f;             // 1/sqrt(32)
    sc[t] = a;
    mx = fmaxf(mx, a);
  }
  float se = 0.f;
  for (int t = 0; t < S_; ++t) { sc[t] = expf(sc[t] - mx); se += sc[t]; }
  float inv = 1.0f / se;

  float o[HEAD_];
  #pragma unroll
  for (int d = 0; d < HEAD_; ++d) o[d] = 0.f;
  for (int t = 0; t < S_; ++t) {
    float w = sc[t] * inv;
    const f16* vp = Vv + ((((long)b * S_ + t) * V_ + vox) * E_ + h * HEAD_);
    #pragma unroll
    for (int d = 0; d < HEAD_; ++d) o[d] += w * (float)vp[d];
  }
  f16* op = O + ((((long)b * S_ + s) * V_ + vox) * E_ + h * HEAD_);
  #pragma unroll
  for (int d = 0; d < HEAD_; ++d) op[d] = (f16)o[d];
}

// ---------------------------------------------------------------------------
// score[b][c][p][vox] = logits[(b*PRED+p)*512+vox][c]
// ---------------------------------------------------------------------------
__global__ void score_k(const float* __restrict__ logits, float* __restrict__ score) {
  long gid = (long)blockIdx.x * blockDim.x + threadIdx.x;
  if (gid >= (long)NMASK_ * NE_) return;
  int c = (int)(gid % NE_);
  long row = gid / NE_;
  int vox = (int)(row & 511);
  int p   = (int)((row >> 9) & 3);
  int b   = (int)(row >> 11);
  score[((((long)b * NE_ + c) * PRED_ + p) << 9) + vox] = logits[gid];
}

__global__ void zero_loss_k(float* p) {
  if (blockIdx.x == 0 && threadIdx.x == 0) p[0] = 0.f;
}

// ---------------------------------------------------------------------------
// log-softmax loss + argmax: one wave32 per masked row
// ---------------------------------------------------------------------------
__global__ void loss_argmax_k(const float* __restrict__ logits,
                              const int* __restrict__ ncode,
                              float* __restrict__ loss, float* __restrict__ code_out) {
  int row  = blockIdx.x * (blockDim.x >> 5) + (threadIdx.x >> 5);
  int lane = threadIdx.x & 31;
  if (row >= NMASK_) return;
  const float* lr = logits + (long)row * NE_;
  float mx = -3.4e38f; int am = 0;
  for (int c = lane; c < NE_; c += 32) {
    float v = lr[c];
    if (v > mx) { mx = v; am = c; }
  }
  #pragma unroll
  for (int o = 16; o; o >>= 1) {
    float om = __shfl_xor(mx, o, 32);
    int   oa = __shfl_xor(am, o, 32);
    if (om > mx || (om == mx && oa < am)) { mx = om; am = oa; }
  }
  float se = 0.f;
  for (int c = lane; c < NE_; c += 32) se += expf(lr[c] - mx);
  #pragma unroll
  for (int o = 16; o; o >>= 1) se += __shfl_xor(se, o, 32);
  if (lane == 0) {
    float lse = mx + logf(se);
    int tgt = ncode[row];
    atomicAdd(loss, (lse - lr[tgt]) * (1.0f / (float)NMASK_));
    code_out[row] = (float)am;
  }
}

// ---------------------------------------------------------------------------
extern "C" void kernel_launch(void* const* d_in, const int* in_sizes, int n_in,
                              void* d_out, int out_size, void* d_ws, size_t ws_size,
                              hipStream_t stream) {
  (void)in_sizes; (void)n_in; (void)out_size; (void)ws_size;
  const int*   code   = (const int*)  d_in[0];
  const int*   ncode  = (const int*)  d_in[1];
  const float* emb    = (const float*)d_in[2];
  const float* pos    = (const float*)d_in[3];
  const float* ln1s   = (const float*)d_in[4];
  const float* ln1b   = (const float*)d_in[5];
  const float* ln2s   = (const float*)d_in[6];
  const float* ln2b   = (const float*)d_in[7];
  const float* wq     = (const float*)d_in[8];
  const float* bq     = (const float*)d_in[9];
  const float* wk     = (const float*)d_in[10];
  const float* bk     = (const float*)d_in[11];
  const float* wv     = (const float*)d_in[12];
  const float* bv     = (const float*)d_in[13];
  const float* wo     = (const float*)d_in[14];
  const float* bo     = (const float*)d_in[15];
  const float* w1     = (const float*)d_in[16];
  const float* b1     = (const float*)d_in[17];
  const float* w2     = (const float*)d_in[18];
  const float* b2     = (const float*)d_in[19];
  const float* lnfs   = (const float*)d_in[20];
  const float* lnfb   = (const float*)d_in[21];
  const float* decw   = (const float*)d_in[22];
  const float* decb   = (const float*)d_in[23];
  const float* declns = (const float*)d_in[24];
  const float* declnb = (const float*)d_in[25];
  const float* linw   = (const float*)d_in[26];
  const float* linb   = (const float*)d_in[27];

  float* out = (float*)d_out;
  float* score_out = out;                                 // [B, NE, PRED, 512]
  float* loss_out  = out + (long)B_ * NE_ * PRED_ * V_;   // 1
  float* code_out  = loss_out + 1;                        // [B, PRED, 512]

  // ---- workspace carve ----
  char* ws = (char*)d_ws;
  size_t off = 0;
  auto take = [&](size_t bytes) -> char* {
    off = (off + 255) & ~(size_t)255;
    char* p = ws + off; off += bytes; return p;
  };
  const size_t SZ_EE  = (size_t)27 * 256 * 256;   // halves
  const size_t SZ_QKV = (size_t)27 * 256 * 768;
  const size_t SZ_EH  = (size_t)27 * 256 * 512;
  const size_t SZ_HE  = (size_t)27 * 512 * 256;

  f16 *wpqkv[L_], *wpo[L_], *wp1[L_], *wp2[L_];
  float* bqkv[L_];
  for (int l = 0; l < L_; ++l) {
    wpqkv[l] = (f16*)take(SZ_QKV * 2);
    wpo[l]   = (f16*)take(SZ_EE * 2);
    wp1[l]   = (f16*)take(SZ_EH * 2);
    wp2[l]   = (f16*)take(SZ_HE * 2);
    bqkv[l]  = (float*)take(768 * 4);
  }
  f16* wpdec = (f16*)take(SZ_EE * 2);
  f16* wplin = (f16*)take((size_t)256 * 512 * 2);

  const long PLANE = (long)NTOK_ * E_;
  float* X     = (float*)take((size_t)NTOK_ * E_ * 4);
  f16*   A16   = (f16*)  take((size_t)NTOK_ * E_ * 2);
  f16*   QKV16 = (f16*)  take((size_t)3 * NTOK_ * E_ * 2);
  f16*   O16   = (f16*)  take((size_t)NTOK_ * E_ * 2);
  f16*   H16   = (f16*)  take((size_t)NTOK_ * HID_ * 2);
  float* DEC32 = (float*)take((size_t)NMASK_ * E_ * 4);
  f16*   Y16   = (f16*)  take((size_t)NMASK_ * E_ * 2);
  float* LOGITS = (float*)take((size_t)NMASK_ * NE_ * 4);

  auto packLaunch = [&](const float* w, f16* wp, int Cin, int Cout, int ntaps) {
    long nd = (long)ntaps * Cin * Cout / 2;
    int grid = (int)((nd + 255) / 256);
    pack_w_k<<<grid, 256, 0, stream>>>(w, wp, Cin, Cout, ntaps);
  };
  auto convLaunch = [&](const f16* in, const f16* wp, const float* bias,
                        float* o32, f16* o16, float* resid,
                        int Cin, int Cout, int ntaps, int M, int mapm, int gelu,
                        long split) {
    int grid = (M / 128) * (Cout / 64);
    conv_wmma_k<<<grid, 128, 0, stream>>>(in, wp, bias, o32, o16, resid,
                                          Cin, Cout, ntaps, M, mapm, gelu, split);
  };

  // ---- weight pre-pack (deterministic, every launch) ----
  for (int l = 0; l < L_; ++l) {
    {
      long nd = (long)SZ_QKV / 2;
      int grid = (int)((nd + 255) / 256);
      pack_qkv_k<<<grid, 256, 0, stream>>>(wq + (size_t)l * SZ_EE,
                                           wk + (size_t)l * SZ_EE,
                                           wv + (size_t)l * SZ_EE, wpqkv[l]);
    }
    packLaunch(wo + (size_t)l * SZ_EE, wpo[l], 256, 256, 27);
    packLaunch(w1 + (size_t)l * SZ_EH, wp1[l], 256, 512, 27);
    packLaunch(w2 + (size_t)l * SZ_HE, wp2[l], 512, 256, 27);
    concat_bias_k<<<3, 256, 0, stream>>>(bq + l * E_, bk + l * E_, bv + l * E_, bqkv[l]);
  }
  packLaunch(decw, wpdec, 256, 256, 27);
  packLaunch(linw, wplin, 256, 512, 1);

  // ---- embedding ----
  {
    long n = (long)NTOK_ * E_;
    embed_k<<<(int)((n + 255) / 256), 256, 0, stream>>>(code, emb, pos, X);
  }

  // ---- transformer layers ----
  const int lnGrid = (NTOK_ + 7) / 8;   // 256 thr = 8 waves = 8 rows/block
  for (int l = 0; l < L_; ++l) {
    layernorm_k<<<lnGrid, 256, 0, stream>>>(X, ln1s + l * E_, ln1b + l * E_, A16, NTOK_, E_);
    convLaunch(A16, wpqkv[l], bqkv[l], nullptr, QKV16, nullptr,
               256, 768, 27, NTOK_, 0, 0, PLANE);
    {
      long n = (long)B_ * V_ * NH_ * S_;
      attention_k<<<(int)((n + 255) / 256), 256, 0, stream>>>(QKV16, PLANE, O16);
    }
    convLaunch(O16, wpo[l], bo + l * E_, nullptr, nullptr, X, 256, 256, 27, NTOK_, 0, 0, 0);
    layernorm_k<<<lnGrid, 256, 0, stream>>>(X, ln2s + l * E_, ln2b + l * E_, A16, NTOK_, E_);
    convLaunch(A16, wp1[l], b1 + l * HID_, nullptr, H16, nullptr, 256, 512, 27, NTOK_, 0, 1, 0);
    convLaunch(H16, wp2[l], b2 + l * E_, nullptr, nullptr, X, 512, 256, 27, NTOK_, 0, 0, 0);
  }

  // ---- decoder (only PRED masked frames) ----
  layernorm_k<<<lnGrid, 256, 0, stream>>>(X, lnfs, lnfb, A16, NTOK_, E_);
  convLaunch(A16, wpdec, decb, DEC32, nullptr, nullptr, 256, 256, 27, NMASK_, 1, 1, 0);
  layernorm_k<<<(NMASK_ + 7) / 8, 256, 0, stream>>>(DEC32, declns, declnb, Y16, NMASK_, E_);
  convLaunch(Y16, wplin, linb, LOGITS, nullptr, nullptr, 256, 512, 1, NMASK_, 0, 0, 0);

  // ---- outputs: score transpose, loss, argmax ----
  {
    long n = (long)NMASK_ * NE_;
    score_k<<<(int)((n + 255) / 256), 256, 0, stream>>>(LOGITS, score_out);
  }
  zero_loss_k<<<1, 32, 0, stream>>>(loss_out);
  loss_argmax_k<<<(NMASK_ + 7) / 8, 256, 0, stream>>>(LOGITS, ncode, loss_out, code_out);
}
